// PatchCritic_67843303407666
// MI455X (gfx1250) — matrix operand order repaired and checked by reading, not compiled, not run on hardware
//
#include <hip/hip_runtime.h>
#include <hip/hip_bf16.h>

// ---------------- problem constants ----------------
#define HW      96
#define CCH     256
#define P       95            // patches per dim (H-1)
#define NREAL   (P * P)       // 9025 patch rows
#define DDIM    1024          // 4*C features per patch
#define NPAD    9088          // 71 * 128 (GEMM tile multiple)
#define TILE    128
#define LDSROW  40            // LDS B-tile row stride in bf16 elems (32 data + 8 pad = 80B)

typedef __attribute__((ext_vector_type(16))) __bf16 v16bf;
typedef __attribute__((ext_vector_type(8)))  float  v8f;
typedef __attribute__((ext_vector_type(4)))  unsigned int u32x4;

union FragBF {
    v16bf bf;
    u32x4 u[2];
};

// ---------------- helpers ----------------

// Patch feature gather: patch n, feature d -> value from [1,256,96,96] tensor.
// d = block*256 + c, block order (di,dj) = (0,0),(0,1),(1,0),(1,1).
__device__ __forceinline__ float patchVal(const float* __restrict__ f, int n, int d) {
    int i   = n / P;
    int j   = n - i * P;
    int blk = d >> 8;
    int c   = d & 255;
    int di  = blk >> 1;
    int dj  = blk & 1;
    return f[(size_t)c * (HW * HW) + (size_t)(i + di) * HW + (j + dj)];
}

__device__ __forceinline__ unsigned short f2bf_rne(float x) {
    unsigned u = __float_as_uint(x);
    unsigned r = u + 0x7FFFu + ((u >> 16) & 1u);
    return (unsigned short)(r >> 16);
}

__device__ __forceinline__ float waveSum32(float v) {
    #pragma unroll
    for (int m = 16; m; m >>= 1) v += __shfl_xor(v, m, 32);
    return v;
}

// monotonic float -> sortable u32 key; ~idx low so max picks smallest index on ties
__device__ __forceinline__ unsigned long long packKeyIdx(float s, unsigned idx) {
    unsigned b   = __float_as_uint(s);
    unsigned key = (b & 0x80000000u) ? ~b : (b | 0x80000000u);
    return ((unsigned long long)key << 32) | (unsigned)(~idx);
}

__device__ __forceinline__ unsigned long long shflXorU64(unsigned long long v, int m) {
    int lo = __shfl_xor((int)(unsigned)v, m, 32);
    int hi = __shfl_xor((int)(unsigned)(v >> 32), m, 32);
    return ((unsigned long long)(unsigned)hi << 32) | (unsigned)lo;
}

// ---------------- kernel 0: zero init ----------------
__global__ void zero_init(unsigned long long* __restrict__ bt,
                          unsigned long long* __restrict__ bs,
                          float* __restrict__ accum) {
    int i = blockIdx.x * blockDim.x + threadIdx.x;
    if (i < NPAD) { bt[i] = 0ull; bs[i] = 0ull; }
    if (i < 2)    accum[i] = 0.0f;
}

// ---------------- kernel 1: build normalized bf16 patch rows ----------------
// one wave32 per padded row; rows >= NREAL are zero-filled (masked later)
__global__ void normalize_pack(const float* __restrict__ feat,
                               unsigned short* __restrict__ dst) {
    int wid  = (int)((blockIdx.x * blockDim.x + threadIdx.x) >> 5);
    int lane = (int)(threadIdx.x & 31);
    if (wid >= NPAD) return;

    unsigned short* row = dst + (size_t)wid * DDIM;
    int base = lane * 32;                     // each lane: 32 consecutive features

    if (wid >= NREAL) {
        u32x4 z = {0u, 0u, 0u, 0u};
        u32x4* p = (u32x4*)(row + base);
        #pragma unroll
        for (int q = 0; q < 4; ++q) p[q] = z; // 32 bf16 = 64B per lane
        return;
    }

    float v[32];
    float ss = 0.0f;
    #pragma unroll
    for (int k = 0; k < 32; ++k) {
        float x = patchVal(feat, wid, base + k);
        v[k] = x;
        ss += x * x;
    }
    ss = waveSum32(ss);
    float inv = 1.0f / (sqrtf(ss) + 1e-8f);

    unsigned packed[16];
    #pragma unroll
    for (int k = 0; k < 16; ++k) {
        unsigned lo = f2bf_rne(v[2 * k]     * inv);
        unsigned hi = f2bf_rne(v[2 * k + 1] * inv);
        packed[k] = lo | (hi << 16);
    }
    u32x4* p = (u32x4*)(row + base);
    #pragma unroll
    for (int q = 0; q < 4; ++q) {
        u32x4 w = { packed[4*q], packed[4*q+1], packed[4*q+2], packed[4*q+3] };
        p[q] = w;
    }
}

// ---------------- kernel 2: bf16 WMMA GEMM + dual argmax epilogue ----------------
// scores[t][s] = tn[t] . sn[s]; block = 128x128 C-tile; wave w owns rows [w*16,w*16+16).
// B-tile (128 cols x 32 k) is staged in LDS once per K-step (double buffered) so
// global/L2 B traffic is shared by all 8 waves (8x reduction vs direct loads).
// All 8 B fragments are loaded into distinct register buffers before the WMMA
// burst so the ds_load latency pipelines behind the matrix ops (partial dscnt waits).
__global__ void __launch_bounds__(256)
gemm_argmax(const unsigned short* __restrict__ tn,
            const unsigned short* __restrict__ sn,
            unsigned long long* __restrict__ best_t,
            unsigned long long* __restrict__ best_s) {
    __shared__ unsigned short ldsB[2][TILE * LDSROW];   // 2 x 10 KB

    const int tid  = (int)threadIdx.x;
    const int lane = tid & 31;
    const int wv   = tid >> 5;                          // 0..7
    const int g    = lane >> 4;                         // half-wave group
    const int nn   = lane & 15;

    const int tRow = (int)blockIdx.y * TILE + wv * 16;  // this wave's 16 target rows
    const int sCol = (int)blockIdx.x * TILE;            // block's 128 source cols

    // A row for this lane (M = nn), contiguous k order matching B:
    //   VGPR0..3 <- K = g*16 + 0..7 ; VGPR4..7 <- K = g*16 + 8..15
    const unsigned short* aRow = tn + (size_t)(tRow + nn) * DDIM + g * 16;

    // B staging: thread pair covers one of 128 rows; each thread copies 16B
    const int stRow  = tid >> 1;                        // 0..127
    const int stHalf = tid & 1;                         // 0 or 1 (k 0..15 / 16..31)
    const unsigned short* stSrc = sn + (size_t)(sCol + stRow) * DDIM + stHalf * 16;
    unsigned short* stDst0 = &ldsB[0][stRow * LDSROW + stHalf * 16];
    unsigned short* stDst1 = &ldsB[1][stRow * LDSROW + stHalf * 16];

    // per-lane LDS read base: B rows (K x 16), lane group g supplies K = g*16+0..15
    // of column nn; sub-tile `sub` starts at row sub*16.
    const unsigned short* bLane0 = &ldsB[0][nn * LDSROW + g * 16];

    v8f acc[8];
    #pragma unroll
    for (int sub = 0; sub < 8; ++sub) acc[sub] = (v8f)(0.0f);

    // prologue: stage k0 = 0 into buffer 0
    *(u32x4*)stDst0 = *(const u32x4*)(stSrc);
    __syncthreads();

    for (int k0 = 0; k0 < DDIM; k0 += 32) {
        const int cur = (k0 >> 5) & 1;
        // stage next K-step into the other buffer (no one reads it until after barrier)
        if (k0 + 32 < DDIM) {
            unsigned short* dst = cur ? stDst0 : stDst1;
            *(u32x4*)dst = *(const u32x4*)(stSrc + (k0 + 32));
        }

        FragBF af;
        af.u[0] = *(const u32x4*)(aRow + k0);          // 8 bf16
        af.u[1] = *(const u32x4*)(aRow + k0 + 8);      // 8 bf16

        // load ALL 8 B fragments into distinct registers first (pipelineable)
        const unsigned short* bLane = bLane0 + cur * (TILE * LDSROW);
        FragBF bfr[8];
        #pragma unroll
        for (int sub = 0; sub < 8; ++sub) {
            const unsigned short* bRow = bLane + sub * (16 * LDSROW);
            bfr[sub].u[0] = *(const u32x4*)(bRow);
            bfr[sub].u[1] = *(const u32x4*)(bRow + 8);
        }
        // WMMA burst
        #pragma unroll
        for (int sub = 0; sub < 8; ++sub) {
            acc[sub] = __builtin_amdgcn_wmma_f32_16x16x32_bf16(
                false, af.bf, false, bfr[sub].bf, (short)0, acc[sub], false, false);
        }
        __syncthreads();
    }

    // C layout: element (m, n): m = r + 8*g (VGPR r), n = lane&15

    // ---- column argmax (best target row per source col), folded across 8 waves via atomics
    #pragma unroll
    for (int sub = 0; sub < 8; ++sub) {
        int col = sCol + sub * 16 + nn;
        unsigned long long cand = 0ull;
        #pragma unroll
        for (int r = 0; r < 8; ++r) {
            int row = tRow + r + 8 * g;
            if (row < NREAL) {
                unsigned long long pk = packKeyIdx(acc[sub][r], (unsigned)row);
                if (pk > cand) cand = pk;
            }
        }
        unsigned long long o = shflXorU64(cand, 16);
        if (o > cand) cand = o;
        if (g == 0 && col < NREAL && cand)
            atomicMax(&best_s[col], cand);
    }

    // ---- row argmax (best source col per target row)
    #pragma unroll
    for (int r = 0; r < 8; ++r) {
        int row = tRow + r + 8 * g;
        unsigned long long cand = 0ull;
        #pragma unroll
        for (int sub = 0; sub < 8; ++sub) {
            int col = sCol + sub * 16 + nn;
            if (col < NREAL) {
                unsigned long long pk = packKeyIdx(acc[sub][r], (unsigned)col);
                if (pk > cand) cand = pk;
            }
        }
        #pragma unroll
        for (int m = 8; m; m >>= 1) {
            unsigned long long o = shflXorU64(cand, m);
            if (o > cand) cand = o;
        }
        if (nn == 0 && row < NREAL && cand)
            atomicMax(&best_t[row], cand);
    }
}

// ---------------- kernel 3: gathered MSE losses (f32, from original features) ----------------
__global__ void loss_kernel(const float* __restrict__ src_feat,
                            const float* __restrict__ tgt_feat,
                            const unsigned long long* __restrict__ best_t,
                            const unsigned long long* __restrict__ best_s,
                            float* __restrict__ accum) {
    int wid  = (int)((blockIdx.x * blockDim.x + threadIdx.x) >> 5);
    int lane = (int)(threadIdx.x & 31);
    if (wid >= 2 * NREAL) return;

    float ss = 0.0f;
    if (wid < NREAL) {
        int t = wid;
        unsigned idx = ~(unsigned)(best_t[t] & 0xFFFFFFFFull); // best source patch
        for (int k = lane; k < DDIM; k += 32) {
            float d = patchVal(tgt_feat, t, k) - patchVal(src_feat, (int)idx, k);
            ss += d * d;
        }
        ss = waveSum32(ss);
        if (lane == 0) atomicAdd(&accum[0], ss);
    } else {
        int s = wid - NREAL;
        unsigned idx = ~(unsigned)(best_s[s] & 0xFFFFFFFFull); // best target patch
        for (int k = lane; k < DDIM; k += 32) {
            float d = patchVal(tgt_feat, (int)idx, k) - patchVal(src_feat, s, k);
            ss += d * d;
        }
        ss = waveSum32(ss);
        if (lane == 0) atomicAdd(&accum[1], ss);
    }
}

// ---------------- kernel 4: finalize ----------------
__global__ void finalize(const float* __restrict__ accum, float* __restrict__ out) {
    if (threadIdx.x == 0 && blockIdx.x == 0) {
        const float invND = 1.0f / ((float)NREAL * (float)DDIM);
        out[0] = 0.5f * accum[0] * invND;   // loss_target
        out[1] = 0.5f * accum[1] * invND;   // loss_source
    }
}

// ---------------- launch ----------------
extern "C" void kernel_launch(void* const* d_in, const int* in_sizes, int n_in,
                              void* d_out, int out_size, void* d_ws, size_t ws_size,
                              hipStream_t stream) {
    const float* src = (const float*)d_in[0];   // source_feat
    const float* tgt = (const float*)d_in[1];   // target_feat
    float* out = (float*)d_out;

    char* ws = (char*)d_ws;
    const size_t matBytes = (size_t)NPAD * DDIM * sizeof(unsigned short); // 18.6 MB
    unsigned short*      tn     = (unsigned short*)ws;
    unsigned short*      snrm   = (unsigned short*)(ws + matBytes);
    unsigned long long*  best_t = (unsigned long long*)(ws + 2 * matBytes);
    unsigned long long*  best_s = best_t + NPAD;
    float*               accum  = (float*)(best_s + NPAD);

    zero_init<<<(NPAD + 255) / 256, 256, 0, stream>>>(best_t, best_s, accum);

    // 8 waves per block, one wave per padded row
    normalize_pack<<<NPAD / 8, 256, 0, stream>>>(tgt, tn);
    normalize_pack<<<NPAD / 8, 256, 0, stream>>>(src, snrm);

    dim3 grid(NPAD / TILE, NPAD / TILE);    // 71 x 71 tiles of 128x128
    gemm_argmax<<<grid, 256, 0, stream>>>(tn, snrm, best_t, best_s);

    int lossBlocks = (2 * NREAL + 7) / 8;   // one wave per row, 8 waves/block
    loss_kernel<<<lossBlocks, 256, 0, stream>>>(src, tgt, best_t, best_s, accum);

    finalize<<<1, 64, 0, stream>>>(accum, out);
}
